// SimpleNodeAggregator_51677046505796
// MI455X (gfx1250) — matrix operand order repaired and checked by reading, not compiled, not run on hardware
//
#include <hip/hip_runtime.h>

#define NN   6144
#define HH   4
#define DD   32
#define JT   256            // j-chunk staged in LDS per iteration
#define TST  272            // transposed LDS tile row stride in bf16 (544B = 17*32B)

typedef __attribute__((ext_vector_type(16))) __bf16 v16bf;
typedef __attribute__((ext_vector_type(8)))  float  v8f;

// ---------------------------------------------------------------------------
// Phase 1: Wh = x @ W per head (fp32 math). Store Wh TRANSPOSED as bf16
// (WhT[h][col][n]) so phase 2 can stage/consume it with wide vector ops.
// Also store packed (exp(e), exp(0.01e)) pairs; later
// exp(lrelu(s+d)) == max(exp(s)exp(d), exp(.01s)exp(.01d)) -> branchless.
// ---------------------------------------------------------------------------
__global__ void gat_phase1(const float* __restrict__ x,
                           const float* __restrict__ W,
                           const float* __restrict__ a_src,
                           const float* __restrict__ a_dst,
                           float2* __restrict__ EsP,      // [H*N] (exp(s), exp(.01s))
                           float2* __restrict__ EdP,      // [H*N] (exp(d), exp(.01d))
                           __bf16* __restrict__ WhT)      // [H*32][N] transposed
{
    __shared__ float xs[128];
    __shared__ float whs[128];
    const int n = blockIdx.x;
    const int t = threadIdx.x;

    xs[t] = x[n * 128 + t];
    __syncthreads();

    const int h = t >> 5, dc = t & 31;
    const float* Wp = W + h * (128 * 32) + dc;   // W[h][f][dc], stride 32 over f
    float acc = 0.f;
#pragma unroll 8
    for (int f = 0; f < 128; ++f) acc = fmaf(xs[f], Wp[f * 32], acc);

    WhT[((size_t)h * 32 + dc) * NN + n] = (__bf16)acc;   // transposed store
    whs[t] = acc;
    __syncthreads();

    if (t < 8) {
        const int hh = t >> 1, which = t & 1;
        const float* av = which ? (a_dst + hh * 32) : (a_src + hh * 32);
        float e = 0.f;
#pragma unroll
        for (int q = 0; q < 32; ++q) e = fmaf(whs[hh * 32 + q], av[q], e);
        const int idx = hh * NN + n;
        const float2 pr = make_float2(expf(e), expf(0.01f * e));
        if (!which) EsP[idx] = pr;
        else        EdP[idx] = pr;
    }
}

// ---------------------------------------------------------------------------
// Phase 2: fused attention @ Wh via v_wmma_f32_16x16x32_bf16.
// WG = 256 threads = 8 waves; one head + 128 output rows per WG,
// one 16-row strip per wave (two 16x16 f32 accumulators -> 16x32 output).
// P[i,j] = max(Es_i*Ed_j, Es01_i*Ed01_j); Z_i accumulated in fp32.
// hp never materialized: relu + column reduction atomically into hsum[128].
// kk loop kept rolled: one iteration fits the VGPR file with zero spills;
// 8 waves/WG provide the latency hiding instead of unrolling.
// ---------------------------------------------------------------------------
__global__ void
__launch_bounds__(256)
__attribute__((amdgpu_waves_per_eu(2, 4)))
gat_phase2(const float2* __restrict__ EsP,
           const float2* __restrict__ EdP,
           const __bf16* __restrict__ WhT,
           float* __restrict__ hsum)
{
    __shared__ float2 ld_e2[JT];                         // (Ed, Ed01) per staged j
    __shared__ __align__(32) __bf16 ld_whT[32 * TST];    // transposed Wh tile [col][j]
    __shared__ float zred[8][32];

    const int blocksPerHead = NN / 128;
    const int h   = blockIdx.x / blocksPerHead;
    const int ib  = (blockIdx.x % blocksPerHead) * 128;
    const int tid = threadIdx.x;
    const int wave = tid >> 5, lane = tid & 31;
    const int m    = lane & 15;          // A-layout row within strip; B/C column n
    const int half = lane >> 4;
    const int row  = ib + wave * 16 + m; // global node index for this lane's A rows

    const float2 es = EsP[h * NN + row];
    const float Es_m = es.x, Es01_m = es.y;

    v8f c0 = {}; v8f c1 = {};
    float zpart = 0.f;

    const int kAbase = half ? 8 : 0;     // A layout: lanes 16-31 hold K 8..15 / 24..31
    const int kBbase = half ? 8 : 0;     // B layout: lanes 16-31 hold K 8..15 / 24..31

    // staging assignment: 8 threads per Wh column, 32 j's (64B) each
    const int sc = tid >> 3;             // column 0..31
    const int sg = tid & 7;              // j-group 0..7

    const float4* e4 = (const float4*)ld_e2;             // two (Ed,Ed01) pairs per load

#pragma clang loop unroll(disable)
    for (int j0 = 0; j0 < NN; j0 += JT) {
        // ---- cooperative staging (coalesced uint4 loads -> b128 LDS stores) ----
        {
            ld_e2[tid] = EdP[h * NN + j0 + tid];
            const uint4* src = (const uint4*)(WhT + ((size_t)h * 32 + sc) * NN + j0 + sg * 32);
            uint4* dst = (uint4*)(ld_whT + sc * TST + sg * 32);
#pragma unroll
            for (int q = 0; q < 4; ++q) dst[q] = src[q];
        }
        __syncthreads();

#pragma clang loop unroll(disable)
        for (int kk = 0; kk < JT; kk += 32) {
            // ---- Build A tile (16x32 bf16, ISA 16-bit A layout) + fp32 Z ----
            v16bf a;
            float z = 0.f;
            {
                const int base0 = (kk + kAbase) >> 1;        // run 1: K = kAbase+0..7
                const int base1 = (kk + kAbase + 16) >> 1;   // run 2: K = kAbase+16..23
#pragma unroll
                for (int t = 0; t < 4; ++t) {
                    const float4 q = e4[base0 + t];
                    const float p0 = fmaxf(Es_m * q.x, Es01_m * q.y);
                    const float p1 = fmaxf(Es_m * q.z, Es01_m * q.w);
                    z += p0 + p1;
                    a[2 * t]     = (__bf16)p0;
                    a[2 * t + 1] = (__bf16)p1;
                }
#pragma unroll
                for (int t = 0; t < 4; ++t) {
                    const float4 q = e4[base1 + t];
                    const float p0 = fmaxf(Es_m * q.x, Es01_m * q.y);
                    const float p1 = fmaxf(Es_m * q.z, Es01_m * q.w);
                    z += p0 + p1;
                    a[8 + 2 * t]     = (__bf16)p0;
                    a[8 + 2 * t + 1] = (__bf16)p1;
                }
            }
            zpart += z;

            // ---- B tiles: two 16B reads (K base+0..7, base+16..23) from transposed tile ----
            union { v16bf v; uint4 q[2]; } ub0, ub1;
            {
                const uint4* p0 = (const uint4*)(ld_whT + m * TST + kk + kBbase);
                const uint4* p1 = (const uint4*)(ld_whT + (m + 16) * TST + kk + kBbase);
                ub0.q[0] = p0[0]; ub0.q[1] = p0[2];
                ub1.q[0] = p1[0]; ub1.q[1] = p1[2];
            }

            c0 = __builtin_amdgcn_wmma_f32_16x16x32_bf16(false, a, false, ub0.v, (short)0, c0, false, false);
            c1 = __builtin_amdgcn_wmma_f32_16x16x32_bf16(false, a, false, ub1.v, (short)0, c1, false, false);
        }
        __syncthreads();
    }

    // Combine the two half-lane Z partials per row
    zred[wave][lane] = zpart;
    __syncthreads();

    // Normalize, relu (relu(elu(x)) == relu(x)), reduce rows, accumulate columns
    float col0 = 0.f, col1 = 0.f;
#pragma unroll
    for (int r = 0; r < 8; ++r) {
        const int M = r + half * 8;                     // C-layout row for VGPR r
        const float invZ = 1.0f / (zred[wave][M] + zred[wave][M + 16]);
        col0 += fmaxf(c0[r] * invZ, 0.f);
        col1 += fmaxf(c1[r] * invZ, 0.f);
    }
    atomicAdd(&hsum[h * 32 + m],      col0);
    atomicAdd(&hsum[h * 32 + 16 + m], col1);
}

// ---------------------------------------------------------------------------
// Phase 3: out = (hsum/N) @ W_fc + b_fc   (128x128, one workgroup)
// ---------------------------------------------------------------------------
__global__ void gat_phase3(const float* __restrict__ hsum,
                           const float* __restrict__ W_fc,
                           const float* __restrict__ b_fc,
                           float* __restrict__ out)
{
    __shared__ float hm[128];
    const int t = threadIdx.x;
    hm[t] = hsum[t] * (1.0f / NN);
    __syncthreads();
    float acc = b_fc[t];
#pragma unroll 8
    for (int k = 0; k < 128; ++k) acc = fmaf(hm[k], W_fc[k * 128 + t], acc);
    out[t] = acc;
}

// ---------------------------------------------------------------------------
extern "C" void kernel_launch(void* const* d_in, const int* in_sizes, int n_in,
                              void* d_out, int out_size, void* d_ws, size_t ws_size,
                              hipStream_t stream)
{
    const float* x     = (const float*)d_in[0];
    const float* W     = (const float*)d_in[1];
    const float* a_src = (const float*)d_in[2];
    const float* a_dst = (const float*)d_in[3];
    const float* W_fc  = (const float*)d_in[4];
    const float* b_fc  = (const float*)d_in[5];
    float* out = (float*)d_out;

    const size_t HN = (size_t)HH * NN;
    char* ws = (char*)d_ws;
    float2* EsP = (float2*)ws; ws += HN * sizeof(float2);
    float2* EdP = (float2*)ws; ws += HN * sizeof(float2);
    __bf16* WhT = (__bf16*)ws; ws += HN * 32 * sizeof(__bf16);
    float*  hsum = (float*)ws; ws += 128 * sizeof(float);

    hipMemsetAsync(hsum, 0, 128 * sizeof(float), stream);

    gat_phase1<<<NN, 128, 0, stream>>>(x, W, a_src, a_dst, EsP, EdP, WhT);

    gat_phase2<<<HH * (NN / 128), 256, 0, stream>>>(EsP, EdP, WhT, hsum);

    gat_phase3<<<1, 128, 0, stream>>>(hsum, W_fc, b_fc, out);
}
